// ResidualBlock_83193516523818
// MI455X (gfx1250) — matrix operand order repaired and checked by reading, not compiled, not run on hardware
//
#include <hip/hip_runtime.h>
#include <math.h>

typedef __attribute__((ext_vector_type(16))) __bf16 v16bf;
typedef __attribute__((ext_vector_type(8)))  __bf16 v8bf;
typedef __attribute__((ext_vector_type(8)))  float  v8f;

#define BATCH   2
#define SEQ     1024
#define DRES    1024
#define DINNER  512
#define NSTATE  256
#define ROWS    (BATCH*SEQ)   /* 2048 */

// ---- fp32 -> bf16 (round to nearest even), packed helpers -----------------
__device__ __forceinline__ unsigned short bfbits(float f) {
  unsigned u = __builtin_bit_cast(unsigned, f);
  u += 0x7FFFu + ((u >> 16) & 1u);
  return (unsigned short)(u >> 16);
}
__device__ __forceinline__ unsigned pack2bf(float lo, float hi) {
  unsigned a = __builtin_bit_cast(unsigned, lo);
  unsigned b = __builtin_bit_cast(unsigned, hi);
  a += 0x7FFFu + ((a >> 16) & 1u);
  b += 0x7FFFu + ((b >> 16) & 1u);
  return (a >> 16) | (b & 0xFFFF0000u);
}
__device__ __forceinline__ float siluf(float x) {
  return x / (1.0f + __expf(-x));
}

// ---- CDNA5 async global->LDS copy (ASYNCcnt-tracked) ----------------------
__device__ __forceinline__ void async_copy_b128(unsigned lds_off, const void* g) {
  asm volatile("global_load_async_to_lds_b128 %0, %1, off"
               :: "v"(lds_off), "v"(g) : "memory");
}
#define WAIT_ASYNC(n) asm volatile("s_wait_asynccnt " #n ::: "memory")

// ---------------------------------------------------------------------------
// LayerNorm: one block (256 threads) per row of 1024; emits bf16 for GEMM A
// ---------------------------------------------------------------------------
__global__ void layernorm_k(const float* __restrict__ x,
                            const float* __restrict__ gamma,
                            const float* __restrict__ beta,
                            unsigned short* __restrict__ ybf) {
  const int row = blockIdx.x;
  const int tid = threadIdx.x;
  const float4 v = ((const float4*)(x + (size_t)row * DRES))[tid];
  float s  = v.x + v.y + v.z + v.w;
  float ss = v.x*v.x + v.y*v.y + v.z*v.z + v.w*v.w;
  #pragma unroll
  for (int m = 16; m >= 1; m >>= 1) {
    s  += __shfl_xor(s,  m, 32);
    ss += __shfl_xor(ss, m, 32);
  }
  __shared__ float sb[8], ssb[8];
  const int wave = tid >> 5, lane = tid & 31;
  if (lane == 0) { sb[wave] = s; ssb[wave] = ss; }
  __syncthreads();
  float ts = 0.f, tss = 0.f;
  #pragma unroll
  for (int i = 0; i < 8; ++i) { ts += sb[i]; tss += ssb[i]; }
  const float mu   = ts * (1.0f / DRES);
  const float var  = tss * (1.0f / DRES) - mu * mu;
  const float rstd = rsqrtf(var + 1e-5f);
  const float4 g = ((const float4*)gamma)[tid];
  const float4 b = ((const float4*)beta)[tid];
  float4 o;
  o.x = (v.x - mu) * rstd * g.x + b.x;
  o.y = (v.y - mu) * rstd * g.y + b.y;
  o.z = (v.z - mu) * rstd * g.z + b.z;
  o.w = (v.w - mu) * rstd * g.w + b.w;
  uint2 p; p.x = pack2bf(o.x, o.y); p.y = pack2bf(o.z, o.w);
  ((uint2*)(ybf + (size_t)row * DRES))[tid] = p;
}

// ---------------------------------------------------------------------------
// One-time weight transpose+convert: W[K][N] fp32 -> Wt[N][K] bf16
// ---------------------------------------------------------------------------
__global__ void w_transpose_bf16(const float* __restrict__ W,
                                 unsigned short* __restrict__ Wt,
                                 int K, int N) {
  __shared__ unsigned short tile[32][33];
  const int kb = blockIdx.x * 32, nb = blockIdx.y * 32;
  const int tx = threadIdx.x & 31;
  const int ty = threadIdx.x >> 5;
  #pragma unroll
  for (int r = ty; r < 32; r += 8)
    tile[tx][r] = bfbits(W[(size_t)(kb + r) * N + nb + tx]);
  __syncthreads();
  #pragma unroll
  for (int r = ty; r < 32; r += 8)
    Wt[(size_t)(nb + r) * K + kb + tx] = tile[r][tx];
}

// ---------------------------------------------------------------------------
// bf16 WMMA GEMM: C[M,N] = act(Abf[M,K] @ Wt[N,K]^T + bias) + resid
// Block tile 128(M) x 64(N), BK=32; 8 waves, each 32x32 (4 accumulators).
// Double-buffered LDS fed by global_load_async_to_lds_b128; software
// pipeline is peeled/unrolled x2 so buffer indices are compile-time.
// ACT: 0=none, 2=softplus
// ---------------------------------------------------------------------------
template<int ACT, bool HAS_BIAS, bool HAS_RES>
__global__ void gemm_bf16_wmma(const unsigned short* __restrict__ Abf,
                               const unsigned short* __restrict__ Wt,
                               const float* __restrict__ bias,
                               const float* __restrict__ resid,
                               float* __restrict__ C,
                               int M, int N, int K) {
  __shared__ alignas(16) unsigned short As[2][128 * 32];
  __shared__ alignas(16) unsigned short Bs[2][64 * 32];

  const int tid  = threadIdx.x;
  const int bm   = blockIdx.x * 128;
  const int bn   = blockIdx.y * 64;
  const int wave = tid >> 5;
  const int lane = tid & 31;
  const int wm   = wave & 3;        // rows wm*32 .. wm*32+31
  const int wn   = wave >> 2;       // cols wn*32 .. wn*32+31

  // staging: thread owns 3 x 16B chunks per K-tile (2 for A, 1 for B)
  const int rS = tid >> 2;              // 0..63
  const int cS = (tid & 3) << 3;        // 0,8,16,24
  const unsigned short* agp0 = Abf + (size_t)(bm + rS) * K + cS;
  const unsigned short* agp1 = Abf + (size_t)(bm + 64 + rS) * K + cS;
  const unsigned short* bgp  = Wt  + (size_t)(bn + rS) * K + cS;
  const unsigned ao0[2] = { (unsigned)(uintptr_t)&As[0][rS * 32 + cS],
                            (unsigned)(uintptr_t)&As[1][rS * 32 + cS] };
  const unsigned ao1[2] = { (unsigned)(uintptr_t)&As[0][(64 + rS) * 32 + cS],
                            (unsigned)(uintptr_t)&As[1][(64 + rS) * 32 + cS] };
  const unsigned bo[2]  = { (unsigned)(uintptr_t)&Bs[0][rS * 32 + cS],
                            (unsigned)(uintptr_t)&Bs[1][rS * 32 + cS] };

  v8f acc00 = {}, acc01 = {}, acc10 = {}, acc11 = {};

  auto issue = [&](int b, int k) {
    async_copy_b128(ao0[b], agp0 + k);
    async_copy_b128(ao1[b], agp1 + k);
    async_copy_b128(bo[b],  bgp  + k);
  };

  auto compute = [&](int b) {
    const __bf16* Ab = (const __bf16*)As[b];
    const __bf16* Bb = (const __bf16*)Bs[b];
    // A fragments: row = lane%16, K(e) = half*8 + (e&7) + (e>=8 ? 16 : 0)
    const int abh  = (lane >> 4) << 3;        // 0 or 8
    const int ar0  = wm * 32 + (lane & 15);
    v8bf a0l = *(const v8bf*)&Ab[ar0 * 32 + abh];
    v8bf a0h = *(const v8bf*)&Ab[ar0 * 32 + 16 + abh];
    v16bf af0 = __builtin_shufflevector(a0l, a0h,
        0,1,2,3,4,5,6,7,8,9,10,11,12,13,14,15);
    const int ar1 = ar0 + 16;
    v8bf a1l = *(const v8bf*)&Ab[ar1 * 32 + abh];
    v8bf a1h = *(const v8bf*)&Ab[ar1 * 32 + 16 + abh];
    v16bf af1 = __builtin_shufflevector(a1l, a1h,
        0,1,2,3,4,5,6,7,8,9,10,11,12,13,14,15);
    // B fragments: col = lane%16, K(e) = half*8 + (e&7) + (e>=8 ? 16 : 0)
    const int kb2 = (lane >> 4) << 3;         // 0 or 8
    const int c0  = wn * 32 + (lane & 15);
    v8bf p0 = *(const v8bf*)&Bb[c0 * 32 + kb2];
    v8bf p1 = *(const v8bf*)&Bb[c0 * 32 + kb2 + 16];
    v16bf bf0 = __builtin_shufflevector(p0, p1,
        0,1,2,3,4,5,6,7,8,9,10,11,12,13,14,15);
    const int c1 = c0 + 16;
    v8bf q0 = *(const v8bf*)&Bb[c1 * 32 + kb2];
    v8bf q1 = *(const v8bf*)&Bb[c1 * 32 + kb2 + 16];
    v16bf bf1 = __builtin_shufflevector(q0, q1,
        0,1,2,3,4,5,6,7,8,9,10,11,12,13,14,15);

    acc00 = __builtin_amdgcn_wmma_f32_16x16x32_bf16(
        false, af0, false, bf0, (short)0, acc00, false, false);
    acc01 = __builtin_amdgcn_wmma_f32_16x16x32_bf16(
        false, af0, false, bf1, (short)0, acc01, false, false);
    acc10 = __builtin_amdgcn_wmma_f32_16x16x32_bf16(
        false, af1, false, bf0, (short)0, acc10, false, false);
    acc11 = __builtin_amdgcn_wmma_f32_16x16x32_bf16(
        false, af1, false, bf1, (short)0, acc11, false, false);
  };

  // ---- software pipeline (K/32 tiles; even count for all our K) ----
  issue(0, 0);
  int k0 = 0;
  for (; k0 + 64 < K; k0 += 64) {
    issue(1, k0 + 32);
    WAIT_ASYNC(0x3);          // tile k0 (buf0) landed
    __syncthreads();
    compute(0);
    __syncthreads();
    issue(0, k0 + 64);
    WAIT_ASYNC(0x3);          // tile k0+32 (buf1) landed
    __syncthreads();
    compute(1);
    __syncthreads();
  }
  // k0 == K-64: tiles k0 (buf0, in flight) and k0+32 remain
  issue(1, k0 + 32);
  WAIT_ASYNC(0x3);
  __syncthreads();
  compute(0);
  __syncthreads();
  WAIT_ASYNC(0x0);
  __syncthreads();
  compute(1);

  // ---- epilogue: D layout -> row = v + 8*(lane>=16), col = lane%16 ----
  const int lhalf = (lane >> 4) << 3;
  const int lcol  = lane & 15;
  #pragma unroll
  for (int mi = 0; mi < 2; ++mi) {
    const int mbase = bm + wm * 32 + mi * 16 + lhalf;
    #pragma unroll
    for (int ni = 0; ni < 2; ++ni) {
      const int nc = bn + wn * 32 + ni * 16 + lcol;
      const float bi = HAS_BIAS ? bias[nc] : 0.f;
      v8f acc = (mi == 0) ? (ni == 0 ? acc00 : acc01)
                          : (ni == 0 ? acc10 : acc11);
      #pragma unroll
      for (int v = 0; v < 8; ++v) {
        const int row = mbase + v;
        float xv = acc[v] + bi;
        if (ACT == 2) xv = (xv > 20.f) ? xv : log1pf(__expf(xv));
        if (HAS_RES)  xv += resid[(size_t)row * N + nc];
        C[(size_t)row * N + nc] = xv;
      }
    }
  }
}

// ---------------------------------------------------------------------------
// Causal depthwise conv (K=3) + SiLU; 4 channels per thread.
// ---------------------------------------------------------------------------
__global__ void dwconv_silu_k(const float* __restrict__ up,
                              const float* __restrict__ w,
                              const float* __restrict__ cbias,
                              float* __restrict__ uf,
                              unsigned short* __restrict__ ubf) {
  const int i   = blockIdx.x * blockDim.x + threadIdx.x;  // over ROWS*DINNER/4
  const int c4  = (i & (DINNER / 4 - 1)) << 2;            // channel base
  const int row = i >> 7;                                 // b*SEQ + t
  const int t   = row & (SEQ - 1);

  float acc[4];
  #pragma unroll
  for (int cc = 0; cc < 4; ++cc) acc[cc] = cbias[c4 + cc];
  #pragma unroll
  for (int j = 0; j < 3; ++j) {
    const int tt = t - 2 + j;
    if (tt >= 0) {
      const float4 v = *(const float4*)(up + (size_t)(row - 2 + j) * (2 * DINNER) + c4);
      acc[0] += w[(c4 + 0) * 3 + j] * v.x;
      acc[1] += w[(c4 + 1) * 3 + j] * v.y;
      acc[2] += w[(c4 + 2) * 3 + j] * v.z;
      acc[3] += w[(c4 + 3) * 3 + j] * v.w;
    }
  }
  float4 o;
  o.x = siluf(acc[0]); o.y = siluf(acc[1]); o.z = siluf(acc[2]); o.w = siluf(acc[3]);
  *(float4*)(uf + (size_t)row * DINNER + c4) = o;
  uint2 p; p.x = pack2bf(o.x, o.y); p.y = pack2bf(o.z, o.w);
  *(uint2*)(ubf + (size_t)row * DINNER + c4) = p;
}

// ---------------------------------------------------------------------------
// S6 selective scan: one wave per (b, d); 8 states per lane (n = lane + 32*i)
// ---------------------------------------------------------------------------
__global__ void s6_scan_k(const float* __restrict__ u,
                          const float* __restrict__ delta,
                          const float* __restrict__ Bm,
                          const float* __restrict__ Cm,
                          const float* __restrict__ A_log,
                          const float* __restrict__ Dp,
                          float* __restrict__ ys) {
  const int wave = threadIdx.x >> 5;
  const int lane = threadIdx.x & 31;
  const int idx  = blockIdx.x * 8 + wave;    // 0 .. BATCH*DINNER-1
  const int b = idx >> 9;
  const int d = idx & (DINNER - 1);

  float a[8], h[8];
  #pragma unroll
  for (int i = 0; i < 8; ++i) {
    const int n = lane + 32 * i;
    a[i] = -__expf(A_log[(size_t)d * NSTATE + n]);
    h[i] = 0.f;
  }
  const float Dd = Dp[d];
  const float* ub = u     + (size_t)b * SEQ * DINNER + d;
  const float* db = delta + (size_t)b * SEQ * DINNER + d;
  const float* Bb = Bm    + (size_t)b * SEQ * NSTATE;
  const float* Cb = Cm    + (size_t)b * SEQ * NSTATE;
  float* yb = ys + (size_t)b * SEQ * DINNER + d;

  for (int t = 0; t < SEQ; ++t) {
    const float dt  = db[(size_t)t * DINNER];
    const float ut  = ub[(size_t)t * DINNER];
    const float dtu = dt * ut;
    float y = 0.f;
    #pragma unroll
    for (int i = 0; i < 8; ++i) {
      const int n = lane + 32 * i;
      const float bt = Bb[(size_t)t * NSTATE + n];
      const float ct = Cb[(size_t)t * NSTATE + n];
      h[i] = __expf(dt * a[i]) * h[i] + dtu * bt;
      y += h[i] * ct;
    }
    #pragma unroll
    for (int m = 16; m >= 1; m >>= 1) y += __shfl_xor(y, m, 32);
    if (lane == 0) yb[(size_t)t * DINNER] = y + ut * Dd;
  }
}

// ---------------------------------------------------------------------------
// gated_bf = bf16( ys * silu(right half of up) ); 4 elements per thread
// ---------------------------------------------------------------------------
__global__ void gate_k(const float* __restrict__ ys,
                       const float* __restrict__ up,
                       unsigned short* __restrict__ gbf) {
  const int i   = blockIdx.x * blockDim.x + threadIdx.x;  // over ROWS*DINNER/4
  const int c4  = (i & (DINNER / 4 - 1)) << 2;
  const int row = i >> 7;
  const float4 yv = *(const float4*)(ys + (size_t)row * DINNER + c4);
  const float4 rv = *(const float4*)(up + (size_t)row * (2 * DINNER) + DINNER + c4);
  float4 g;
  g.x = yv.x * siluf(rv.x);
  g.y = yv.y * siluf(rv.y);
  g.z = yv.z * siluf(rv.z);
  g.w = yv.w * siluf(rv.w);
  uint2 p; p.x = pack2bf(g.x, g.y); p.y = pack2bf(g.z, g.w);
  *(uint2*)(gbf + (size_t)row * DINNER + c4) = p;
}

// ---------------------------------------------------------------------------
extern "C" void kernel_launch(void* const* d_in, const int* in_sizes, int n_in,
                              void* d_out, int out_size, void* d_ws, size_t ws_size,
                              hipStream_t stream) {
  const float* x        = (const float*)d_in[0];
  const float* ln_gamma = (const float*)d_in[1];
  const float* ln_beta  = (const float*)d_in[2];
  const float* W_up     = (const float*)d_in[3];
  const float* b_up     = (const float*)d_in[4];
  const float* conv_w   = (const float*)d_in[5];
  const float* conv_b   = (const float*)d_in[6];
  const float* W_delta  = (const float*)d_in[7];
  const float* b_delta  = (const float*)d_in[8];
  const float* W_B      = (const float*)d_in[9];
  const float* W_C      = (const float*)d_in[10];
  const float* A_log    = (const float*)d_in[11];
  const float* Dvec     = (const float*)d_in[12];
  const float* W_down   = (const float*)d_in[13];
  const float* b_down   = (const float*)d_in[14];
  float* out = (float*)d_out;

  // fp32 scratch
  float* wsf = (float*)d_ws;
  float* up    = wsf;  wsf += (size_t)ROWS * 2 * DINNER;  // 2048x1024
  float* uf    = wsf;  wsf += (size_t)ROWS * DINNER;      // conv+silu (scan input)
  float* dl    = wsf;  wsf += (size_t)ROWS * DINNER;      // softplus(delta)
  float* Bm    = wsf;  wsf += (size_t)ROWS * NSTATE;
  float* Cm    = wsf;  wsf += (size_t)ROWS * NSTATE;
  float* yscan = wsf;  wsf += (size_t)ROWS * DINNER;
  // bf16 scratch
  unsigned short* wsh = (unsigned short*)wsf;
  unsigned short* ln_ybf   = wsh;  wsh += (size_t)ROWS * DRES;
  unsigned short* ubf      = wsh;  wsh += (size_t)ROWS * DINNER;
  unsigned short* gbf      = wsh;  wsh += (size_t)ROWS * DINNER;
  unsigned short* Wt_up    = wsh;  wsh += (size_t)DRES * 2 * DINNER;
  unsigned short* Wt_delta = wsh;  wsh += (size_t)DINNER * DINNER;
  unsigned short* Wt_B     = wsh;  wsh += (size_t)DINNER * NSTATE;
  unsigned short* Wt_C     = wsh;  wsh += (size_t)DINNER * NSTATE;
  unsigned short* Wt_down  = wsh;  wsh += (size_t)DINNER * DRES;

  // 0) weight transpose+convert to [N][K] bf16
  w_transpose_bf16<<<dim3(DRES/32, (2*DINNER)/32), 256, 0, stream>>>(W_up,    Wt_up,    DRES,   2*DINNER);
  w_transpose_bf16<<<dim3(DINNER/32, DINNER/32),   256, 0, stream>>>(W_delta, Wt_delta, DINNER, DINNER);
  w_transpose_bf16<<<dim3(DINNER/32, NSTATE/32),   256, 0, stream>>>(W_B,     Wt_B,     DINNER, NSTATE);
  w_transpose_bf16<<<dim3(DINNER/32, NSTATE/32),   256, 0, stream>>>(W_C,     Wt_C,     DINNER, NSTATE);
  w_transpose_bf16<<<dim3(DINNER/32, DRES/32),     256, 0, stream>>>(W_down,  Wt_down,  DINNER, DRES);

  // 1) LayerNorm -> bf16
  layernorm_k<<<ROWS, 256, 0, stream>>>(x, ln_gamma, ln_beta, ln_ybf);

  // 2) up = ln_y @ W_up + b_up   (2048 x 1024 x 1024)
  gemm_bf16_wmma<0, true, false><<<dim3(ROWS/128, (2*DINNER)/64), 256, 0, stream>>>(
      ln_ybf, Wt_up, b_up, nullptr, up, ROWS, 2*DINNER, DRES);

  // 3) u = silu(causal_dwconv(left))  (fp32 + bf16)
  dwconv_silu_k<<<(ROWS * DINNER / 4) / 256, 256, 0, stream>>>(up, conv_w, conv_b, uf, ubf);

  // 4) delta = softplus(u @ W_delta + b_delta)
  gemm_bf16_wmma<2, true, false><<<dim3(ROWS/128, DINNER/64), 256, 0, stream>>>(
      ubf, Wt_delta, b_delta, nullptr, dl, ROWS, DINNER, DINNER);

  // 5) Bm = u @ W_B ; Cm = u @ W_C
  gemm_bf16_wmma<0, false, false><<<dim3(ROWS/128, NSTATE/64), 256, 0, stream>>>(
      ubf, Wt_B, nullptr, nullptr, Bm, ROWS, NSTATE, DINNER);
  gemm_bf16_wmma<0, false, false><<<dim3(ROWS/128, NSTATE/64), 256, 0, stream>>>(
      ubf, Wt_C, nullptr, nullptr, Cm, ROWS, NSTATE, DINNER);

  // 6) selective scan: 1024 waves, 8 per block
  s6_scan_k<<<(BATCH * DINNER) / 8, 256, 0, stream>>>(uf, dl, Bm, Cm, A_log, Dvec, yscan);

  // 7) gated = yscan * silu(right) -> bf16
  gate_k<<<(ROWS * DINNER / 4) / 256, 256, 0, stream>>>(yscan, up, gbf);

  // 8) out = x + gated @ W_down + b_down
  gemm_bf16_wmma<0, true, true><<<dim3(ROWS/128, DRES/64), 256, 0, stream>>>(
      gbf, Wt_down, b_down, x, out, ROWS, DRES, DINNER);
}
